// STGEmbMamba_51264729645754
// MI455X (gfx1250) — matrix operand order repaired and checked by reading, not compiled, not run on hardware
//
#include <hip/hip_runtime.h>
#include <hip/hip_bf16.h>
#include <math.h>

// ---- dims (match reference) ----
#define BB   4
#define TT   12
#define NN   307
#define EE   96          // embed / d_model
#define DMs  96
#define DSs  64
#define DCs  4
#define DIs  192         // d_inner
#define DTRs 6           // dt_rank
#define NLs  2
#define LL   (TT*NN)     // 3684
#define MM   (BB*LL)     // 14736
#define XPN  134         // DTR + 2*DS
#define XPNP 144         // padded to multiple of 16

typedef __attribute__((ext_vector_type(16))) _Float16 v16h;
typedef __attribute__((ext_vector_type(8)))  float    v8f;

__device__ __forceinline__ float siluf(float x) {
    return x / (1.0f + __expf(-x));
}

// ---------------------------------------------------------------------------
// GAT + positional embedding.  One block per (b,t); emb[b, t*N+n, :] written.
// h = x * gat_W (rank-1), so attention logits collapse to scalars per node.
// ---------------------------------------------------------------------------
__global__ void gat_kernel(const float* __restrict__ x,     // [B,T,N]
                           const float* __restrict__ A,     // [N,N]
                           const float* __restrict__ gW,    // [1,E]
                           const float* __restrict__ aS,    // [E]
                           const float* __restrict__ aD,    // [E]
                           const float* __restrict__ gB,    // [E]
                           const float* __restrict__ pe,    // [T,E]
                           float* __restrict__ emb)         // [B, L, E]
{
    __shared__ float sx[NN];
    const int bt = blockIdx.x;
    const int b  = bt / TT;
    const int t  = bt % TT;
    const int tid = threadIdx.x;

    for (int j = tid; j < NN; j += blockDim.x)
        sx[j] = x[(b*TT + t)*NN + j];

    float ws = 0.f, wd = 0.f;
    for (int e = 0; e < EE; ++e) { ws += gW[e]*aS[e]; wd += gW[e]*aD[e]; }
    __syncthreads();

    for (int i = tid; i < NN; i += blockDim.x) {
        const float si = ws * sx[i];
        float mx = -3.4e38f;
        for (int j = 0; j < NN; ++j) {
            float e = si + wd * sx[j];
            e = (e >= 0.f) ? e : 0.2f * e;               // leaky_relu 0.2
            const bool adj = (A[i*NN + j] > 0.5f) || (i == j);
            e = adj ? e : -1e9f;
            mx = fmaxf(mx, e);
        }
        float ssum = 0.f, r = 0.f;
        for (int j = 0; j < NN; ++j) {
            float e = si + wd * sx[j];
            e = (e >= 0.f) ? e : 0.2f * e;
            const bool adj = (A[i*NN + j] > 0.5f) || (i == j);
            e = adj ? e : -1e9f;
            const float p = __expf(e - mx);
            ssum += p;
            r    += p * sx[j];
        }
        r /= ssum;
        const size_t base = ((size_t)b*LL + (size_t)t*NN + i) * EE;
        for (int e = 0; e < EE; ++e) {
            float v = r * gW[e] + gB[e];
            v = (v > 0.f) ? v : (__expf(v) - 1.f);        // elu
            emb[base + e] = v + pe[t*EE + e];
        }
    }
}

// ---------------------------------------------------------------------------
// Pack fp32 weights [K, Nact] into transposed, zero-padded f16 [Npad, K].
// Makes the WMMA B-fragment a single contiguous 32B aligned load.
// ---------------------------------------------------------------------------
__global__ void pack_w_kernel(const float* __restrict__ W, // [K, Nact]
                              _Float16* __restrict__ Wt,   // [Npad, K]
                              int K, int Nact, int Npad)
{
    const int idx = blockIdx.x * blockDim.x + threadIdx.x;
    if (idx >= Npad * K) return;
    const int n = idx / K;
    const int k = idx % K;
    Wt[idx] = (n < Nact) ? (_Float16)W[(size_t)k * Nact + n] : (_Float16)0.0f;
}

// ---------------------------------------------------------------------------
// WMMA GEMM: Out[M,Npad] = X[M,K] @ Wt^T  (Wt pre-packed f16 [Npad,K]).
// One wave computes a 16 x (16*NT) strip: A fragment loaded once per k-chunk,
// reused across NT B tiles.  K % 32 == 0 (templated, fully unrolled).
// ---------------------------------------------------------------------------
template<int K, int NT>
__global__ void gemm_wmma_kernel(const float* __restrict__ X,
                                 const _Float16* __restrict__ Wt,
                                 float* __restrict__ Out,
                                 int Npad, int Mrows)
{
    const int wave    = threadIdx.x >> 5;
    const int lane    = threadIdx.x & 31;
    const int groupsN = (Npad >> 4) / NT;
    const int group   = blockIdx.x * 4 + wave;           // wave-uniform
    if (group >= (Mrows >> 4) * groupsN) return;         // whole-wave exit

    const int tm  = group / groupsN;
    const int tn0 = (group % groupsN) * NT;

    const int m   = tm*16 + (lane & 15);
    const int hi  = lane >> 4;
    const int col = lane & 15;

    v8f c[NT];
    #pragma unroll
    for (int nt = 0; nt < NT; ++nt) c[nt] = (v8f){};

    #pragma unroll
    for (int k0 = 0; k0 < K; k0 += 32) {
        // A fragment: rows m, K-slice [k0, k0+32); f32 -> f16 convert
        v16h a;
        const size_t xrow = (size_t)m * K + k0;
        #pragma unroll
        for (int j = 0; j < 8; ++j) {
            a[j]     = (_Float16)X[xrow + hi*16 + j];
            a[8 + j] = (_Float16)X[xrow + hi*16 + 8 + j];
        }
        #pragma unroll
        for (int nt = 0; nt < NT; ++nt) {
            const int gcol = (tn0 + nt)*16 + col;
            // contiguous 16 halfs, 32B aligned (K mult of 32, k0+hi*16 mult of 16)
            const v16h bf = *reinterpret_cast<const v16h*>(
                Wt + (size_t)gcol * K + k0 + hi*16);
            c[nt] = __builtin_amdgcn_wmma_f32_16x16x32_f16(
                        false, a, false, bf, (short)0, c[nt], false, false);
        }
    }
    #pragma unroll
    for (int nt = 0; nt < NT; ++nt) {
        #pragma unroll
        for (int r = 0; r < 8; ++r) {
            const int row = tm*16 + r + hi*8;
            Out[(size_t)row * Npad + (tn0 + nt)*16 + col] = c[nt][r];
        }
    }
}

// ---------------------------------------------------------------------------
// Causal depthwise conv (K=4) + SiLU.  xi = xz[:, :, 0:DI] (stride 2*DI).
// ---------------------------------------------------------------------------
__global__ void conv_silu_kernel(const float* __restrict__ xz,
                                 const float* __restrict__ cw, // [DI,4]
                                 const float* __restrict__ cb, // [DI]
                                 float* __restrict__ xc)       // [M,DI]
{
    const size_t idx = (size_t)blockIdx.x * blockDim.x + threadIdx.x;
    if (idx >= (size_t)MM * DIs) return;
    const int c = idx % DIs;
    const int r = idx / DIs;
    const int b = r / LL;
    const int l = r % LL;
    float acc = cb[c];
    #pragma unroll
    for (int k = 0; k < DCs; ++k) {
        const int ls = l - (DCs - 1) + k;
        if (ls >= 0)
            acc += xz[((size_t)b*LL + ls)*(2*DIs) + c] * cw[c*DCs + k];
    }
    xc[idx] = siluf(acc);
}

// ---------------------------------------------------------------------------
// dt = softplus(proj[:, :DTR] @ dtW + dtb)
// ---------------------------------------------------------------------------
__global__ void dt_kernel(const float* __restrict__ proj,  // [M, XPNP]
                          const float* __restrict__ dtW,   // [DTR, DI]
                          const float* __restrict__ dtb,   // [DI]
                          float* __restrict__ dt)          // [M, DI]
{
    const size_t idx = (size_t)blockIdx.x * blockDim.x + threadIdx.x;
    if (idx >= (size_t)MM * DIs) return;
    const int c = idx % DIs;
    const size_t row = idx / DIs;
    float s = dtb[c];
    #pragma unroll
    for (int r = 0; r < DTRs; ++r)
        s += proj[row*XPNP + r] * dtW[r*DIs + c];
    dt[idx] = fmaxf(s, 0.f) + log1pf(__expf(-fabsf(s)));   // stable softplus
}

// ---------------------------------------------------------------------------
// Selective scan.  One block per batch, one thread per channel; 64-state in
// registers; B_t/C_t staged through LDS each step.  Fuses D*u skip and SiLU(z)
// gating; writes y ready for out_proj.
// ---------------------------------------------------------------------------
__global__ void __launch_bounds__(DIs)
scan_kernel(const float* __restrict__ xc,    // [M,DI]
            const float* __restrict__ dt,    // [M,DI]
            const float* __restrict__ proj,  // [M,XPNP] (B at 6, C at 70)
            const float* __restrict__ A_log, // [DI,DS]
            const float* __restrict__ Dp,    // [DI]
            const float* __restrict__ xz,    // [M,2*DI] (z at +DI)
            float* __restrict__ y)           // [M,DI]
{
    __shared__ float sB[DSs];
    __shared__ float sC[DSs];
    const int b = blockIdx.x;
    const int c = threadIdx.x;

    float h[DSs];
    float Am[DSs];
    #pragma unroll
    for (int n = 0; n < DSs; ++n) {
        Am[n] = -__expf(A_log[c*DSs + n]);
        h[n]  = 0.f;
    }
    const float dpc = Dp[c];

    for (int l = 0; l < LL; ++l) {
        const size_t row = (size_t)b*LL + l;
        if (threadIdx.x < DSs)
            sB[threadIdx.x] = proj[row*XPNP + DTRs + threadIdx.x];
        else if (threadIdx.x < 2*DSs)
            sC[threadIdx.x - DSs] = proj[row*XPNP + DTRs + DSs + (threadIdx.x - DSs)];
        __syncthreads();

        const float u  = xc[row*DIs + c];
        const float d  = dt[row*DIs + c];
        const float du = d * u;
        float acc = 0.f;
        #pragma unroll
        for (int n = 0; n < DSs; ++n) {
            const float dA = __expf(d * Am[n]);
            h[n] = dA * h[n] + du * sB[n];
            acc += h[n] * sC[n];
        }
        const float zv = xz[row*(2*DIs) + DIs + c];
        y[row*DIs + c] = (acc + u * dpc) * siluf(zv);
        __syncthreads();
    }
}

// ---------------------------------------------------------------------------
// Head + torch-faithful permute: out[b,t',n'] = head(emb[b, n'*T + t'])
// ---------------------------------------------------------------------------
__global__ void head_kernel(const float* __restrict__ emb,   // [B,L,E]
                            const float* __restrict__ hW,    // [E,1]
                            const float* __restrict__ hb,    // [1]
                            float* __restrict__ out)         // [B,T,N]
{
    const int idx = blockIdx.x * blockDim.x + threadIdx.x;
    if (idx >= BB*TT*NN) return;
    const int b  = idx / (TT*NN);
    const int rr = idx % (TT*NN);
    const int tp = rr / NN;
    const int np = rr % NN;
    const int l  = np*TT + tp;                 // reshape(B,N,T).transpose
    const size_t base = ((size_t)b*LL + l)*EE;
    float acc = hb[0];
    for (int e = 0; e < EE; ++e) acc += emb[base + e] * hW[e];
    out[idx] = acc;
}

// ---------------------------------------------------------------------------
extern "C" void kernel_launch(void* const* d_in, const int* in_sizes, int n_in,
                              void* d_out, int out_size, void* d_ws, size_t ws_size,
                              hipStream_t stream)
{
    const float* inputs  = (const float*)d_in[0];
    const float* A       = (const float*)d_in[1];
    const float* gat_W   = (const float*)d_in[2];
    const float* gat_aS  = (const float*)d_in[3];
    const float* gat_aD  = (const float*)d_in[4];
    const float* gat_b   = (const float*)d_in[5];
    const float* pe      = (const float*)d_in[6];
    const float* in_W    = (const float*)d_in[7];   // [NL,96,384]
    const float* conv_W  = (const float*)d_in[8];   // [NL,192,4]
    const float* conv_b  = (const float*)d_in[9];   // [NL,192]
    const float* xp_W    = (const float*)d_in[10];  // [NL,192,134]
    const float* dt_W    = (const float*)d_in[11];  // [NL,6,192]
    const float* dt_b    = (const float*)d_in[12];  // [NL,192]
    const float* A_log   = (const float*)d_in[13];  // [NL,192,64]
    const float* Dp      = (const float*)d_in[14];  // [NL,192]
    const float* out_W   = (const float*)d_in[15];  // [NL,192,96]
    const float* head_W  = (const float*)d_in[16];  // [96,1]
    const float* head_b  = (const float*)d_in[17];  // [1]
    float* out = (float*)d_out;

    // workspace layout (floats)
    float* ws = (float*)d_ws;
    float* emb  = ws;                              // M*96
    float* xz   = emb  + (size_t)MM*EE;            // M*384
    float* xc   = xz   + (size_t)MM*(2*DIs);       // M*192
    float* proj = xc   + (size_t)MM*DIs;           // M*144
    float* dt   = proj + (size_t)MM*XPNP;          // M*192
    float* yb   = dt   + (size_t)MM*DIs;           // M*192
    _Float16* Wt = (_Float16*)(yb + (size_t)MM*DIs); // up to 384*96 / 144*192 halves

    // 1) GAT + PE embedding
    gat_kernel<<<BB*TT, 256, 0, stream>>>(inputs, A, gat_W, gat_aS, gat_aD,
                                          gat_b, pe, emb);

    const int elemN   = MM * DIs;
    const int elemGrd = (elemN + 255) / 256;

    for (int l = 0; l < NLs; ++l) {
        // ---- in_proj: [M,96]@[96,384] -> xz  (24 n-tiles, NT=4 -> 6 groups)
        pack_w_kernel<<<(2*DIs*DMs + 255)/256, 256, 0, stream>>>(
            in_W + (size_t)l*DMs*2*DIs, Wt, DMs, 2*DIs, 2*DIs);
        {
            const int groups = (MM/16) * ((2*DIs/16)/4);
            gemm_wmma_kernel<DMs, 4><<<(groups + 3)/4, 128, 0, stream>>>(
                emb, Wt, xz, 2*DIs, MM);
        }
        // ---- depthwise causal conv + silu
        conv_silu_kernel<<<elemGrd, 256, 0, stream>>>(
            xz, conv_W + (size_t)l*DIs*DCs, conv_b + (size_t)l*DIs, xc);
        // ---- x_proj: [M,192]@[192,134] -> proj (padded 144; 9 n-tiles, NT=3)
        pack_w_kernel<<<(XPNP*DIs + 255)/256, 256, 0, stream>>>(
            xp_W + (size_t)l*DIs*XPN, Wt, DIs, XPN, XPNP);
        {
            const int groups = (MM/16) * ((XPNP/16)/3);
            gemm_wmma_kernel<DIs, 3><<<(groups + 3)/4, 128, 0, stream>>>(
                xc, Wt, proj, XPNP, MM);
        }
        // ---- dt = softplus(proj[:, :6] @ dtW + dtb)
        dt_kernel<<<elemGrd, 256, 0, stream>>>(
            proj, dt_W + (size_t)l*DTRs*DIs, dt_b + (size_t)l*DIs, dt);
        // ---- selective scan + D*u + silu(z) gating
        scan_kernel<<<BB, DIs, 0, stream>>>(
            xc, dt, proj, A_log + (size_t)l*DIs*DSs, Dp + (size_t)l*DIs, xz, yb);
        // ---- out_proj: [M,192]@[192,96] -> emb  (6 n-tiles, NT=3)
        pack_w_kernel<<<(DMs*DIs + 255)/256, 256, 0, stream>>>(
            out_W + (size_t)l*DIs*DMs, Wt, DIs, DMs, DMs);
        {
            const int groups = (MM/16) * ((DMs/16)/3);
            gemm_wmma_kernel<DIs, 3><<<(groups + 3)/4, 128, 0, stream>>>(
                yb, Wt, emb, DMs, MM);
        }
    }

    // head + output permute
    head_kernel<<<(BB*TT*NN + 255)/256, 256, 0, stream>>>(emb, head_W, head_b, out);
}
